// VQCodebookSeg_multihead_62302795596234
// MI455X (gfx1250) — matrix operand order repaired and checked by reading, not compiled, not run on hardware
//
#include <hip/hip_runtime.h>
#include <hip/hip_bf16.h>

// Shapes from the reference
#define Bq   16
#define Sq   2048
#define Cq   256
#define QKq  256
#define Hq   8
#define Nq   512
#define EDq  256
#define Dq   32           // QK / H
#define SCALINGq 0.0625f  // QK^-0.5

typedef __attribute__((ext_vector_type(16))) __bf16 v16bf;
typedef __attribute__((ext_vector_type(8)))  float  v8f;
typedef __attribute__((ext_vector_type(4)))  float  v4f;

// A-operand (16x32 bf16) in this code's k order (B uses the same), one contiguous run per lane:
//   e in [0,16) -> K = 16*half + e
// so A can be fetched as 4x float4 from rows of a row-major matrix.
__device__ __forceinline__ v16bf load_a_row(const float* __restrict__ rowp, int half) {
    const float* p = rowp + (half << 4);
    const v4f x0 = *(const v4f*)(p + 0);
    const v4f x1 = *(const v4f*)(p + 4);
    const v4f x2 = *(const v4f*)(p + 8);
    const v4f x3 = *(const v4f*)(p + 12);
    v16bf a;
#pragma unroll
    for (int e = 0; e < 4; ++e) {
        a[e]      = (__bf16)x0[e];
        a[e + 4]  = (__bf16)x1[e];
        a[e + 8]  = (__bf16)x2[e];
        a[e + 12] = (__bf16)x3[e];
    }
    return a;
}

// ---------------------------------------------------------------------------
// Kernel 1: kbuf[n][qk] = env_codebook[n,:] @ Wk[:,qk]   (512x256, K=256)
// One wave per 16x16 output tile; 8 bf16 WMMAs per wave.
// ---------------------------------------------------------------------------
__global__ void kproj_kernel(const float* __restrict__ cb,
                             const float* __restrict__ Wk,
                             float* __restrict__ kbuf)
{
    const int wave = threadIdx.x >> 5;
    const int lane = threadIdx.x & 31;
    const int tile = blockIdx.x * 8 + wave;      // 512 tiles total
    const int n0   = (tile >> 4) << 4;           // codebook-row tile base
    const int qk0  = (tile & 15) << 4;           // output-col tile base
    const int m    = lane & 15;
    const int half = lane >> 4;

    v8f acc = {};
    for (int k0 = 0; k0 < EDq; k0 += 32) {
        const v16bf a = load_a_row(cb + (size_t)(n0 + m) * EDq + k0, half);
        v16bf b;
#pragma unroll
        for (int e = 0; e < 16; ++e) {
            const int kb = (half << 4) + e;              // B: K = 16*half + e
            b[e] = (__bf16)Wk[(k0 + kb) * QKq + qk0 + m];
        }
        acc = __builtin_amdgcn_wmma_f32_16x16x32_bf16(
                  false, a, false, b, (short)0, acc, false, false);
    }
#pragma unroll
    for (int r = 0; r < 8; ++r) {
        const int row = n0 + r + (half << 3);
        kbuf[row * QKq + qk0 + m] = acc[r];
    }
}

// ---------------------------------------------------------------------------
// Kernel 2: fused q-projection + scores + softmax + outputs.
// Block = (s_chunk, h, b), 256 threads = 8 waves, wave w owns rows
// s0 = s_chunk*128 + w*16. Outputs written exactly once (store-bound) with
// non-temporal hints so the 512MB env_selects stream doesn't thrash L2.
// ---------------------------------------------------------------------------
__global__ void attn_kernel(const float* __restrict__ x,
                            const float* __restrict__ Wq,
                            const float* __restrict__ kbuf,
                            float* __restrict__ out)
{
    // k head-slice staged in WMMA B-operand layout:
    //   k_bl[nt][lane][e] = bf16( k[nt*16 + (lane&15)][16*(lane>>4) + e] )
    __shared__ __align__(32) __bf16 k_bl[32][32][16];     // 32 KB
    // Wq head-slice staged in B-operand layout, per K-step and N-half:
    //   wq_bl[kt][nh][lane][e] = bf16( Wq[kt*32 + 16*(lane>>4) + e][h*32 + nh*16 + (lane&15)] )
    __shared__ __align__(32) __bf16 wq_bl[8][2][32][16];  // 16 KB
    // per-wave q re-layout buffer (C/D layout -> A layout), bf16
    __shared__ __align__(32) __bf16 q_sh[8][32][16];      // 8 KB

    const int tid  = threadIdx.x;
    const int sc   = blockIdx.x;       // 0..15  (S/128)
    const int h    = blockIdx.y;       // 0..7
    const int b    = blockIdx.z;       // 0..15
    const int wave = tid >> 5;
    const int lane = tid & 31;
    const int m    = lane & 15;
    const int half = lane >> 4;

    // ---- stage k (head slice) into B-operand layout, float4 granularity ----
    for (int i = tid; i < 32 * 32 * 4; i += 256) {
        const int nt = i >> 7;
        const int l  = (i >> 2) & 31;
        const int e4 = (i & 3) << 2;
        const int n  = (nt << 4) + (l & 15);
        const int d  = ((l >> 4) << 4) + e4;
        const v4f kv = *(const v4f*)&kbuf[n * QKq + h * Dq + d];
#pragma unroll
        for (int c = 0; c < 4; ++c) k_bl[nt][l][e4 + c] = (__bf16)kv[c];
    }
    // ---- stage Wq head slice into B-operand layout ----
    for (int i = tid; i < 8 * 2 * 32 * 16; i += 256) {
        const int kt = i >> 10;
        const int nh = (i >> 9) & 1;
        const int l  = (i >> 4) & 31;
        const int e  = i & 15;
        const int k  = (kt << 5) + ((l >> 4) << 4) + e;
        const int col = h * Dq + (nh << 4) + (l & 15);
        wq_bl[kt][nh][l][e] = (__bf16)Wq[k * QKq + col];
    }
    __syncthreads();

    // ---- q-projection: q_tile[16][32] = x_tile[16][256] @ Wq[:, h*32 .. +32) ----
    const int s0 = sc * 128 + wave * 16;
    const float* xrow = x + (((size_t)b * Sq + s0 + m) * Cq);

    v8f qacc0 = {}, qacc1 = {};
#pragma unroll
    for (int kt = 0; kt < 8; ++kt) {
        const v16bf a  = load_a_row(xrow + (kt << 5), half);
        const v16bf b0 = *(const v16bf*)&wq_bl[kt][0][lane][0];
        const v16bf b1 = *(const v16bf*)&wq_bl[kt][1][lane][0];
        qacc0 = __builtin_amdgcn_wmma_f32_16x16x32_bf16(
                    false, a, false, b0, (short)0, qacc0, false, false);
        qacc1 = __builtin_amdgcn_wmma_f32_16x16x32_bf16(
                    false, a, false, b1, (short)0, qacc1, false, false);
    }

    // ---- re-layout q from C/D layout into A-operand layout via LDS ----
#pragma unroll
    for (int nh = 0; nh < 2; ++nh) {
#pragma unroll
        for (int r = 0; r < 8; ++r) {
            const int mm  = r + (half << 3);       // q row 0..15
            const int d   = (nh << 4) + m;         // q col 0..31
            const int dh  = d >> 4;                // target lane half
            const int e   = d & 15;
            const float qv = (nh == 0) ? qacc0[r] : qacc1[r];
            q_sh[wave][mm | (dh << 4)][e] = (__bf16)qv;
        }
    }
    __syncthreads();
    const v16bf qa = *(const v16bf*)&q_sh[wave][lane][0];

    // ---- pass 1: online (max, sum) over the 512 columns, flash-style ----
    float M[8], S[8];
#pragma unroll
    for (int r = 0; r < 8; ++r) { M[r] = -1e30f; S[r] = 0.0f; }

    for (int nt = 0; nt < 32; ++nt) {
        const v16bf bm = *(const v16bf*)&k_bl[nt][lane][0];
        v8f scr = {};
        scr = __builtin_amdgcn_wmma_f32_16x16x32_bf16(
                  false, qa, false, bm, (short)0, scr, false, false);
#pragma unroll
        for (int r = 0; r < 8; ++r) {
            const float v  = scr[r] * SCALINGq;
            const float nm = fmaxf(M[r], v);
            S[r] = S[r] * __expf(M[r] - nm) + __expf(v - nm);
            M[r] = nm;
        }
    }
    // combine across the 16 lanes of each half (columns of a row)
#pragma unroll
    for (int mask = 1; mask <= 8; mask <<= 1) {
#pragma unroll
        for (int r = 0; r < 8; ++r) {
            const float Mo = __shfl_xor(M[r], mask, 32);
            const float So = __shfl_xor(S[r], mask, 32);
            const float nm = fmaxf(M[r], Mo);
            S[r] = S[r] * __expf(M[r] - nm) + So * __expf(Mo - nm);
            M[r] = nm;
        }
    }
    float invS[8];
#pragma unroll
    for (int r = 0; r < 8; ++r) invS[r] = 1.0f / S[r];

    // ---- pass 2: recompute scores, normalize, stream outputs (NT stores) ----
    float* out_env    = out;                            // [B*H*S]
    float* out_causal = out + (size_t)Bq * Hq * Sq;     // [B*H*S]
    float* out_sel    = out + (size_t)2 * Bq * Hq * Sq; // [B*H*S][511]
    const size_t rowbase = (size_t)((b * Hq + h) * Sq + s0);

    for (int nt = 0; nt < 32; ++nt) {
        const v16bf bm = *(const v16bf*)&k_bl[nt][lane][0];
        v8f scr = {};
        scr = __builtin_amdgcn_wmma_f32_16x16x32_bf16(
                  false, qa, false, bm, (short)0, scr, false, false);
        const int n = (nt << 4) + m;     // global column 0..511
#pragma unroll
        for (int r = 0; r < 8; ++r) {
            const int row = r + (half << 3);
            const float p = __expf(scr[r] * SCALINGq - M[r]) * invS[r];
            if (n >= 1) {
                __builtin_nontemporal_store(p, &out_sel[(rowbase + row) * 511 + (n - 1)]);
            } else {
                __builtin_nontemporal_store(p,        &out_causal[rowbase + row]);
                __builtin_nontemporal_store(1.0f - p, &out_env[rowbase + row]);
            }
        }
    }
}

extern "C" void kernel_launch(void* const* d_in, const int* in_sizes, int n_in,
                              void* d_out, int out_size, void* d_ws, size_t ws_size,
                              hipStream_t stream) {
    const float* x   = (const float*)d_in[0];   // (16, 2048, 256)
    const float* cb  = (const float*)d_in[1];   // (512, 256)
    const float* Wq  = (const float*)d_in[2];   // (256, 256)
    const float* Wk  = (const float*)d_in[3];   // (256, 256)
    float* out  = (float*)d_out;
    float* kbuf = (float*)d_ws;                 // 512*256 f32 = 512 KB scratch

    // k projection: 512 16x16 tiles, 8 waves/block -> 64 blocks
    kproj_kernel<<<64, 256, 0, stream>>>(cb, Wk, kbuf);

    // fused attention: grid = (S/128, H, B)
    dim3 grid(Sq / 128, Hq, Bq);
    attn_kernel<<<grid, 256, 0, stream>>>(x, Wq, kbuf, out);
}
